// GatedDeltaNet_23673859735746
// MI455X (gfx1250) — compile-verified, matrix operand order checked by reading, not run on hardware
//
#include <hip/hip_runtime.h>
#include <hip/hip_bf16.h>
#include <cstdint>
#include <cstddef>

// ---------------------------------------------------------------------------
// GatedDeltaNet forward for MI455X (gfx1250, wave32, WMMA, async/TDM).
//  - Projection GEMMs: bf16, v_wmma_f32_16x16x32_bf16, fp32 accum,
//    double-buffered LDS staging via GLOBAL_LOAD_ASYNC_TO_LDS_B128 (ASYNCcnt).
//  - Delta-rule core:  fp32, v_wmma_f32_16x16x4_f32, 128x128 state in LDS,
//    next-chunk q/k/v prefetch via TDM tensor_load_to_lds (TENSORcnt).
// ---------------------------------------------------------------------------

typedef float v2f  __attribute__((ext_vector_type(2)));
typedef float v8f  __attribute__((ext_vector_type(8)));
typedef __bf16 bf16x16 __attribute__((ext_vector_type(16)));
typedef unsigned int u32x4 __attribute__((ext_vector_type(4)));
typedef int i32x4 __attribute__((ext_vector_type(4)));
typedef int i32x8 __attribute__((ext_vector_type(8)));

#if __has_builtin(__builtin_amdgcn_global_load_async_to_lds_b128)
#define HAS_ASYNC_LDS 1
#else
#define HAS_ASYNC_LDS 0
#endif
#if __has_builtin(__builtin_amdgcn_tensor_load_to_lds)
#define HAS_TDM 1
#else
#define HAS_TDM 0
#endif

static constexpr int Bb    = 2;
static constexpr int Tt    = 4096;
static constexpr int Dd    = 2048;
static constexpr int HK    = 16;
static constexpr int HVn   = 32;
static constexpr int AK    = 128;
static constexpr int UV    = 128;
static constexpr int KEYD  = HK * AK;          // 2048
static constexpr int VALD  = HVn * UV;         // 4096
static constexpr int CONVD = 2 * KEYD + VALD;  // 8192
static constexpr int MTOT  = Bb * Tt;          // 8192
static constexpr int CH    = 64;               // chunk
static constexpr int NJ    = Tt / CH;          // 64 chunks

__device__ inline v8f v8zero() {
  v8f x;
#pragma unroll
  for (int i = 0; i < 8; ++i) x[i] = 0.f;
  return x;
}

__device__ inline void wait_asynccnt0() {
#if __has_builtin(__builtin_amdgcn_s_wait_asynccnt)
  __builtin_amdgcn_s_wait_asynccnt(0);
#else
  asm volatile("s_wait_asynccnt 0" ::: "memory");
#endif
}

// async global->LDS 16B copy (per-lane); ASYNCcnt-tracked.
__device__ inline void async_copy_b128(const void* gsrc, void* ldst) {
#if HAS_ASYNC_LDS
  __builtin_amdgcn_global_load_async_to_lds_b128(
      (__attribute__((address_space(1))) i32x4*)gsrc,
      (__attribute__((address_space(3))) i32x4*)ldst, 0, 0);
#else
  (void)gsrc; (void)ldst;
#endif
}

// Issue one TDM 2D tile load: rows x cols fp32, row stride in elements,
// global tile start -> LDS byte offset. D# per CDNA5 ISA §8 (groups 0/1;
// groups 2/3 zero => dims 2..4 unused).
__device__ inline void tdm_load_2d_f32(unsigned lds_byte_off, const float* gsrc,
                                       int rows, int cols, int row_stride_elems) {
#if HAS_TDM
  unsigned long long ga = (unsigned long long)(const void*)gsrc;
  u32x4 g0;
  g0.x = 1u;                                                  // count=1, user D#
  g0.y = lds_byte_off;                                        // lds_addr
  g0.z = (unsigned)(ga & 0xFFFFFFFFull);                      // global_addr lo
  g0.w = (unsigned)((ga >> 32) & 0x01FFFFFFull) | 0x80000000u;// addr hi | type=2
  i32x8 g1;
  g1[0] = (int)(2u << 16);                  // data_size=4B; mask=0; no pad/iter
  g1[1] = (int)((unsigned)cols << 16);      // tensor_dim0[15:0]
  g1[2] = (int)((unsigned)rows << 16);      // tensor_dim0 hi=0 | tensor_dim1 lo
  g1[3] = (int)((unsigned)cols << 16);      // tensor_dim1 hi=0 | tile_dim0
  g1[4] = rows;                             // tile_dim1 | tile_dim2=0
  g1[5] = row_stride_elems;                 // tensor_dim0_stride lo
  g1[6] = 0;                                // stride hi | dim1_stride lo
  g1[7] = 0;
  i32x4 z4 = {0, 0, 0, 0};
#if defined(__clang_major__) && __clang_major__ >= 23
  i32x8 z8 = {0, 0, 0, 0, 0, 0, 0, 0};
  __builtin_amdgcn_tensor_load_to_lds(g0, g1, z4, z4, z8, 0);
#else
  __builtin_amdgcn_tensor_load_to_lds(g0, g1, z4, z4, 0);
#endif
#else
  (void)lds_byte_off; (void)gsrc; (void)rows; (void)cols; (void)row_stride_elems;
#endif
}

__device__ inline unsigned lds_off_of(const void* p) {
  // generic pointer to __shared__: low 32 bits are the LDS byte offset
  return (unsigned)(unsigned long long)p;
}

// ---------------------------------------------------------------------------
// fp32 -> bf16 cast
// ---------------------------------------------------------------------------
__global__ void k_cast_bf16(const float* __restrict__ s, __bf16* __restrict__ d, long n) {
  long i = (long)blockIdx.x * blockDim.x + threadIdx.x;
  if (i < n) d[i] = (__bf16)s[i];
}

// ---------------------------------------------------------------------------
// bf16 GEMM: C(MxN fp32) = A(MxK bf16, row-major) @ B(KxN bf16, row-major)
// 128x128 block, BK=32, 8 waves (2x4), each wave 64x32 = 4x2 WMMA tiles.
// Double-buffered LDS, staged with async global->LDS DMA when available.
// ---------------------------------------------------------------------------
__global__ __launch_bounds__(256) void k_gemm_bf16(
    const __bf16* __restrict__ A, const __bf16* __restrict__ Bm,
    float* __restrict__ C, int M, int N, int K) {
  __shared__ __align__(16) __bf16 sA[2][128 * 32];
  __shared__ __align__(16) __bf16 sB[2][32 * 128];
  const int tid  = threadIdx.x;
  const int wave = tid >> 5, lane = tid & 31;
  const int col  = lane & 15, half = lane >> 4;
  const int wm = wave >> 2, wn = wave & 3;
  const int m0 = blockIdx.y * 128, n0 = blockIdx.x * 128;

  v8f acc[4][2];
#pragma unroll
  for (int i = 0; i < 4; ++i)
#pragma unroll
    for (int j = 0; j < 2; ++j) acc[i][j] = v8zero();

#if HAS_ASYNC_LDS
  auto stage_async = [&](int b, int k0) {
#pragma unroll
    for (int c = tid; c < 512; c += 256) {        // A tile: 512 x 16B
      int row = c >> 2, seg = c & 3;
      async_copy_b128(A + (size_t)(m0 + row) * K + k0 + seg * 8,
                      &sA[b][row * 32 + seg * 8]);
    }
#pragma unroll
    for (int c = tid; c < 512; c += 256) {        // B tile: 512 x 16B
      int row = c >> 4, seg = c & 15;
      async_copy_b128(Bm + (size_t)(k0 + row) * N + n0 + seg * 8,
                      &sB[b][row * 128 + seg * 8]);
    }
  };
#else
  auto stage_sync = [&](int b, int k0) {
    {  // A tile (uint4 = 8 bf16)
      int row = tid >> 1, seg = tid & 1;
      const uint4* src = (const uint4*)(A + (size_t)(m0 + row) * K + k0 + seg * 16);
      uint4* dst = (uint4*)(&sA[b][row * 32 + seg * 16]);
      dst[0] = src[0]; dst[1] = src[1];
    }
    {  // B tile
      int row = tid >> 3, seg = tid & 7;
      const uint4* src = (const uint4*)(Bm + (size_t)(k0 + row) * N + n0 + seg * 16);
      uint4* dst = (uint4*)(&sB[b][row * 128 + seg * 16]);
      dst[0] = src[0]; dst[1] = src[1];
    }
  };
#endif

  auto compute = [&](int b) {
#pragma unroll
    for (int i = 0; i < 4; ++i) {
      bf16x16 a;
      int ar = wm * 64 + i * 16 + col;
#pragma unroll
      for (int e = 0; e < 16; ++e) {  // ISA 16-bit A 16x32 layout
        int v = e >> 1, bit = e & 1;
        int kk = (v < 4) ? (8 * half + 2 * v + bit) : (16 + 8 * half + 2 * (v - 4) + bit);
        a[e] = sA[b][ar * 32 + kk];
      }
#pragma unroll
      for (int j = 0; j < 2; ++j) {
        bf16x16 bb;
        int bc = wn * 32 + j * 16 + col;
#pragma unroll
        for (int e = 0; e < 16; ++e) {
          int v = e >> 1, bit = e & 1;
          int kk = (v < 4) ? (8 * half + 2 * v + bit) : (16 + 8 * half + 2 * (v - 4) + bit);
          bb[e] = sB[b][kk * 128 + bc];
        }
        acc[i][j] = __builtin_amdgcn_wmma_f32_16x16x32_bf16(
            false, a, false, bb, (short)0, acc[i][j], false, false);
      }
    }
  };

  int buf = 0;
#if HAS_ASYNC_LDS
  stage_async(0, 0);
  wait_asynccnt0();
  __syncthreads();
#endif
  for (int k0 = 0; k0 < K; k0 += 32) {
#if HAS_ASYNC_LDS
    if (k0 + 32 < K) stage_async(buf ^ 1, k0 + 32);  // DMA overlaps WMMA below
    compute(buf);
    wait_asynccnt0();
    __syncthreads();
    buf ^= 1;
#else
    stage_sync(buf, k0);
    __syncthreads();
    compute(buf);
    __syncthreads();
#endif
  }

#pragma unroll
  for (int i = 0; i < 4; ++i)
#pragma unroll
    for (int j = 0; j < 2; ++j)
#pragma unroll
      for (int e = 0; e < 8; ++e) {
        int r = m0 + wm * 64 + i * 16 + e + half * 8;
        int c = n0 + wn * 32 + j * 16 + col;
        C[(size_t)r * N + c] = acc[i][j][e];
      }
}

// ---------------------------------------------------------------------------
// b/a projections (N=32 each, too small for the 128x128 GEMM): simple dot.
// ---------------------------------------------------------------------------
__global__ void k_ba_proj(const float* __restrict__ hidden, const float* __restrict__ Wb,
                          const float* __restrict__ Wa, float* __restrict__ bp,
                          float* __restrict__ ap) {
  long i = (long)blockIdx.x * 256 + threadIdx.x;
  if (i >= (long)MTOT * 64) return;
  int o = (int)(i & 63);
  long m = i >> 6;
  const float* hrow = hidden + m * Dd;
  const float* w = (o < 32) ? (Wb + o) : (Wa + (o - 32));
  float acc = 0.f;
  for (int kk = 0; kk < Dd; ++kk) acc += hrow[kk] * w[(size_t)kk * 32];
  if (o < 32) bp[m * 32 + o] = acc;
  else        ap[m * 32 + (o - 32)] = acc;
}

// ---------------------------------------------------------------------------
// causal depthwise conv (K=4) + silu, split into q/k/v buffers.
// ---------------------------------------------------------------------------
__global__ void k_conv_silu(const float* __restrict__ mixed, const float* __restrict__ cw,
                            float* __restrict__ q, float* __restrict__ k,
                            float* __restrict__ v) {
  long i = (long)blockIdx.x * 256 + threadIdx.x;
  if (i >= (long)MTOT * CONVD) return;
  int c  = (int)(i % CONVD);
  long m = i / CONVD;
  int t  = (int)(m % Tt);
  float acc = cw[c * 4 + 3] * mixed[(size_t)m * CONVD + c];
  if (t >= 1) acc += cw[c * 4 + 2] * mixed[(size_t)(m - 1) * CONVD + c];
  if (t >= 2) acc += cw[c * 4 + 1] * mixed[(size_t)(m - 2) * CONVD + c];
  if (t >= 3) acc += cw[c * 4 + 0] * mixed[(size_t)(m - 3) * CONVD + c];
  float s = acc / (1.f + expf(-acc));  // silu
  if (c < KEYD)          q[(size_t)m * KEYD + c] = s;
  else if (c < 2 * KEYD) k[(size_t)m * KEYD + (c - KEYD)] = s;
  else                   v[(size_t)m * VALD + (c - 2 * KEYD)] = s;
}

// ---------------------------------------------------------------------------
// l2-norm of q,k per (row, head) over A=128; q additionally * A^-0.5.
// ---------------------------------------------------------------------------
__global__ void k_qk_norm(float* __restrict__ q, float* __restrict__ k) {
  int wave = threadIdx.x >> 5, lane = threadIdx.x & 31;
  long task = (long)blockIdx.x * 8 + wave;
  const long nq = (long)MTOT * HK;
  float* p;
  float scale;
  if (task < nq) { p = q + task * AK; scale = 0.08838834764831845f; }  // 128^-0.5
  else {
    long t2 = task - nq;
    if (t2 >= nq) return;
    p = k + t2 * AK; scale = 1.f;
  }
  float x[4]; float ss = 0.f;
#pragma unroll
  for (int e = 0; e < 4; ++e) { x[e] = p[lane * 4 + e]; ss += x[e] * x[e]; }
#pragma unroll
  for (int off = 16; off >= 1; off >>= 1) ss += __shfl_xor(ss, off, 32);
  float inv = rsqrtf(ss + 1e-6f) * scale;
#pragma unroll
  for (int e = 0; e < 4; ++e) p[lane * 4 + e] = x[e] * inv;
}

// ---------------------------------------------------------------------------
// beta = sigmoid(b), g = -exp(A_log) * softplus(a + dt_bias)
// ---------------------------------------------------------------------------
__global__ void k_beta_g(const float* __restrict__ bp, const float* __restrict__ ap,
                         const float* __restrict__ dtb, const float* __restrict__ Alog,
                         float* __restrict__ beta, float* __restrict__ g) {
  long i = (long)blockIdx.x * 256 + threadIdx.x;
  if (i >= (long)MTOT * HVn) return;
  int h = (int)(i % HVn);
  beta[i] = 1.f / (1.f + expf(-bp[i]));
  float x = ap[i] + dtb[h];
  float sp = (x > 20.f) ? x : log1pf(expf(x));
  g[i] = -expf(Alog[h]) * sp;
}

// ---------------------------------------------------------------------------
// fp32 WMMA tiled matmul over LDS operands (delta-core helper).
//   Dst[M x N] (ld ldd)  (+)=  sgn * (A' @ B')
//   A'[m][k] = A[(m0+m)*lda_r + k*lda_k] * (arow ? arow[m0+m] : 1)
//   B'[k][n] = B[k*ldb_k + (n0+n)*ldb_n] * (bks ? bks[k] : 1)
// EXEC stays full (wave-uniform control) as WMMA requires.
// ---------------------------------------------------------------------------
__device__ inline void mm_f32(float* __restrict__ Dst, int ldd,
                              const float* __restrict__ A, int lda_r, int lda_k,
                              const float* __restrict__ arow,
                              const float* __restrict__ Bm, int ldb_k, int ldb_n,
                              const float* __restrict__ bks,
                              int M, int N, int K, float sgn, bool acc) {
  const int wave = threadIdx.x >> 5, lane = threadIdx.x & 31;
  const int r = lane & 15, half = lane >> 4;
  const int nt = N >> 4;
  const int tiles = (M >> 4) * nt;
  for (int t = wave; t < tiles; t += 8) {
    const int m0 = (t / nt) << 4, n0 = (t % nt) << 4;
    v8f c;
    if (acc) {
#pragma unroll
      for (int e = 0; e < 8; ++e) c[e] = Dst[(m0 + e + half * 8) * ldd + n0 + r];
    } else {
      c = v8zero();
    }
    for (int k0 = 0; k0 < K; k0 += 4) {
      const int ka = k0 + half * 2;  // ISA 32-bit A 16x4: K = 2*half + elem
      float fa = sgn * (arow ? arow[m0 + r] : 1.f);
      v2f a, b;
      a.x = A[(m0 + r) * lda_r + ka * lda_k] * fa;
      a.y = A[(m0 + r) * lda_r + (ka + 1) * lda_k] * fa;
      float f0 = bks ? bks[ka] : 1.f;
      float f1 = bks ? bks[ka + 1] : 1.f;
      b.x = Bm[ka * ldb_k + (n0 + r) * ldb_n] * f0;
      b.y = Bm[(ka + 1) * ldb_k + (n0 + r) * ldb_n] * f1;
#if __has_builtin(__builtin_amdgcn_wmma_f32_16x16x4_f32)
      c = __builtin_amdgcn_wmma_f32_16x16x4_f32(false, a, false, b, (short)0, c,
                                                false, false);
#else
#pragma unroll
      for (int e = 0; e < 8; ++e) c[e] += a.x * b.x + a.y * b.y;
#endif
    }
#pragma unroll
    for (int e = 0; e < 8; ++e) Dst[(m0 + e + half * 8) * ldd + n0 + r] = c[e];
  }
}

// ---------------------------------------------------------------------------
// Chunked gated delta rule. One block per (batch, head) stream; 64 sequential
// chunks of 64 timesteps. 128x128 fp32 state + double-buffered q/k/v tiles in
// LDS (~313 KB of the 320 KB WGP budget). Next chunk's q/k/v are prefetched by
// the Tensor Data Mover while the current chunk's WMMAs run.
// ---------------------------------------------------------------------------
__global__ __launch_bounds__(256) void k_delta_core(
    const float* __restrict__ qb, const float* __restrict__ kb,
    const float* __restrict__ vb, const float* __restrict__ beta,
    const float* __restrict__ g, float* __restrict__ core) {
  __shared__ __align__(16) float s_S[128 * 128];
  __shared__ __align__(16) float s_q[2][64 * 128];
  __shared__ __align__(16) float s_k[2][64 * 128];
  __shared__ __align__(16) float s_v[2][64 * 128];
  __shared__ __align__(16) float s_tmp[64 * 128];
  __shared__ __align__(16) float s_T[64 * 64];
  __shared__ float s_gc[64], s_beta[64], s_eg[64], s_egr[64], s_be[64];
  __shared__ float s_red[4 * 64];

  const int tid = threadIdx.x;
  const int bh = blockIdx.x;
  const int bbatch = bh >> 5;  // / HV
  const int h = bh & 31;
  const int hk = h >> 1;       // GQA: repeat factor 2

  for (int i = tid; i < 128 * 128; i += 256) s_S[i] = 0.f;
  __syncthreads();

  int buf = 0;
#if HAS_TDM
  if (tid < 32) {  // wave 0 issues TDM descriptors for chunk 0
    const long mb0 = (long)bbatch * Tt;
    tdm_load_2d_f32(lds_off_of(&s_q[0][0]), qb + mb0 * KEYD + hk * AK, 64, 128, KEYD);
    tdm_load_2d_f32(lds_off_of(&s_k[0][0]), kb + mb0 * KEYD + hk * AK, 64, 128, KEYD);
    tdm_load_2d_f32(lds_off_of(&s_v[0][0]), vb + mb0 * VALD + h * UV, 64, 128, VALD);
  }
#endif

  for (int j = 0; j < NJ; ++j) {
    const long mbase = (long)bbatch * Tt + j * CH;
    float* sq = s_q[buf];
    float* sk = s_k[buf];
    float* sv = s_v[buf];

    if (tid < 64) {
      s_beta[tid] = beta[(mbase + tid) * HVn + h];
      s_gc[tid]   = g[(mbase + tid) * HVn + h];
    }
#if HAS_TDM
    if (tid < 32) __builtin_amdgcn_s_wait_tensorcnt(0);  // chunk j tiles landed
#else
    {
      const float* qsrc = qb + mbase * KEYD + hk * AK;
      const float* ksrc = kb + mbase * KEYD + hk * AK;
      const float* vsrc = vb + mbase * VALD + h * UV;
      for (int i = tid; i < 64 * 32; i += 256) {
        int l = i >> 5, c4 = i & 31;
        ((float4*)sq)[i] = *(const float4*)(qsrc + (long)l * KEYD + c4 * 4);
        ((float4*)sk)[i] = *(const float4*)(ksrc + (long)l * KEYD + c4 * 4);
        ((float4*)sv)[i] = *(const float4*)(vsrc + (long)l * VALD + c4 * 4);
      }
    }
#endif
    __syncthreads();

#if HAS_TDM
    if (j + 1 < NJ && tid < 32) {  // prefetch chunk j+1 while computing chunk j
      const long mbn = mbase + CH;
      tdm_load_2d_f32(lds_off_of(&s_q[buf ^ 1][0]), qb + mbn * KEYD + hk * AK, 64, 128, KEYD);
      tdm_load_2d_f32(lds_off_of(&s_k[buf ^ 1][0]), kb + mbn * KEYD + hk * AK, 64, 128, KEYD);
      tdm_load_2d_f32(lds_off_of(&s_v[buf ^ 1][0]), vb + mbn * VALD + h * UV, 64, 128, VALD);
    }
#endif

    if (tid == 0)
      for (int i = 1; i < 64; ++i) s_gc[i] += s_gc[i - 1];  // in-chunk cumsum
    __syncthreads();
    if (tid < 64) {
      float gi = s_gc[tid];
      s_eg[tid]  = expf(gi);
      s_egr[tid] = expf(s_gc[63] - gi);
      s_be[tid]  = s_beta[tid] * s_eg[tid];
    }
    __syncthreads();

    // (1) raw P = (k*beta) @ k^T, then attn = -P * decay, strictly lower.
    mm_f32(s_T, 64, sk, 128, 1, s_beta, sk, 1, 128, nullptr, 64, 64, 128, 1.f, false);
    __syncthreads();
    for (int i = tid; i < 64 * 64; i += 256) {
      int rr = i >> 6, cc = i & 63;
      s_T[i] = (rr > cc) ? -expf(s_gc[rr] - s_gc[cc]) * s_T[i] : 0.f;
    }
    __syncthreads();

    // (2) forward substitution: row_i += row_i @ T (i = 1..63), then +I.
    for (int i = 1; i < 64; ++i) {
      int cc = tid & 63, p = tid >> 6;
      float part = 0.f;
      int j0 = p * 16;
      int j1 = (j0 + 16 > i) ? i : (j0 + 16);
      for (int jj = j0; jj < j1; ++jj) part += s_T[i * 64 + jj] * s_T[jj * 64 + cc];
      s_red[p * 64 + cc] = part;
      __syncthreads();
      if (p == 0)
        s_T[i * 64 + cc] += s_red[cc] + s_red[64 + cc] + s_red[128 + cc] + s_red[192 + cc];
      __syncthreads();
    }
    if (tid < 64) s_T[tid * 65] = 1.f;
    __syncthreads();

    // (3) s_tmp = (k*beta*e^g) @ S - (v*beta)    [associativity avoids k_cd]
    for (int i = tid; i < 64 * 128; i += 256) {
      int rr = i >> 7;
      s_tmp[i] = -s_beta[rr] * sv[i];
    }
    __syncthreads();
    mm_f32(s_tmp, 128, sk, 128, 1, s_be, s_S, 128, 1, nullptr, 64, 128, 128, 1.f, true);
    __syncthreads();

    // (4) v_new = -(T @ s_tmp)  -> overwrite sv
    mm_f32(sv, 128, s_T, 64, 1, nullptr, s_tmp, 128, 1, nullptr, 64, 128, 64, -1.f, false);
    __syncthreads();

    // (5) intra = (q @ k^T) * decay, lower incl. diag -> reuse s_T
    mm_f32(s_T, 64, sq, 128, 1, nullptr, sk, 1, 128, nullptr, 64, 64, 128, 1.f, false);
    __syncthreads();
    for (int i = tid; i < 64 * 64; i += 256) {
      int rr = i >> 6, cc = i & 63;
      s_T[i] = (rr >= cc) ? s_T[i] * expf(s_gc[rr] - s_gc[cc]) : 0.f;
    }
    __syncthreads();
    //     out = (q*e^g) @ S + intra @ v_new
    mm_f32(s_tmp, 128, sq, 128, 1, s_eg, s_S, 128, 1, nullptr, 64, 128, 128, 1.f, false);
    __syncthreads();
    mm_f32(s_tmp, 128, s_T, 64, 1, nullptr, sv, 128, 1, nullptr, 64, 128, 64, 1.f, true);
    __syncthreads();
    {
      float* dst = core + mbase * VALD + h * UV;
      for (int i = tid; i < 64 * 32; i += 256) {
        int l = i >> 5, c4 = i & 31;
        *(float4*)(dst + (long)l * VALD + c4 * 4) = ((float4*)s_tmp)[i];
      }
    }

    // (6) S = S * e^{g_last} + k^T @ (v_new * e^{g_last - g})
    float eglast = s_eg[63];
    __syncthreads();
    for (int i = tid; i < 128 * 128; i += 256) s_S[i] *= eglast;
    __syncthreads();
    mm_f32(s_S, 128, sk, 1, 128, nullptr, sv, 128, 1, s_egr, 128, 128, 64, 1.f, true);
    __syncthreads();

    buf ^= 1;
  }
}

// ---------------------------------------------------------------------------
// xg = core * silu(z); RMSNorm over UV=128 group; * norm_w; emit bf16.
// ---------------------------------------------------------------------------
__global__ void k_gate_norm(const float* __restrict__ core, const float* __restrict__ z,
                            const float* __restrict__ nw, __bf16* __restrict__ out) {
  int wave = threadIdx.x >> 5, lane = threadIdx.x & 31;
  long task = (long)blockIdx.x * 8 + wave;
  if (task >= (long)MTOT * HVn) return;
  const float* xc = core + task * UV;
  const float* zc = z + task * UV;
  float x[4]; float ss = 0.f;
#pragma unroll
  for (int e = 0; e < 4; ++e) {
    float zz = zc[lane * 4 + e];
    float xv = xc[lane * 4 + e] * (zz / (1.f + expf(-zz)));
    x[e] = xv; ss += xv * xv;
  }
#pragma unroll
  for (int off = 16; off >= 1; off >>= 1) ss += __shfl_xor(ss, off, 32);
  float inv = rsqrtf(ss * (1.f / 128.f) + 1e-6f);
#pragma unroll
  for (int e = 0; e < 4; ++e)
    out[task * UV + lane * 4 + e] = (__bf16)(x[e] * inv * nw[lane * 4 + e]);
}

// ---------------------------------------------------------------------------
extern "C" void kernel_launch(void* const* d_in, const int* in_sizes, int n_in,
                              void* d_out, int out_size, void* d_ws, size_t ws_size,
                              hipStream_t stream) {
  (void)in_sizes; (void)n_in; (void)out_size; (void)ws_size;
  const float* hidden = (const float*)d_in[0];
  const float* Wqkv   = (const float*)d_in[1];
  const float* Wz     = (const float*)d_in[2];
  const float* Wb     = (const float*)d_in[3];
  const float* Wa     = (const float*)d_in[4];
  const float* cw     = (const float*)d_in[5];
  const float* dtb    = (const float*)d_in[6];
  const float* Alog   = (const float*)d_in[7];
  const float* nw     = (const float*)d_in[8];
  const float* Wout   = (const float*)d_in[9];
  float* out = (float*)d_out;

  char* ws = (char*)d_ws;
  size_t off = 0;
  auto alloc = [&](size_t bytes) {
    void* p = ws + off;
    off += (bytes + 255) & ~(size_t)255;
    return p;
  };
  float* mixed   = (float*)alloc((size_t)MTOT * CONVD * 4);
  float* zbuf    = (float*)alloc((size_t)MTOT * VALD * 4);
  float* qbuf    = (float*)alloc((size_t)MTOT * KEYD * 4);
  float* kbuf    = (float*)alloc((size_t)MTOT * KEYD * 4);
  float* vbuf    = (float*)alloc((size_t)MTOT * VALD * 4);
  float* corebuf = (float*)alloc((size_t)MTOT * VALD * 4);
  float* bp      = (float*)alloc((size_t)MTOT * HVn * 4);
  float* ap      = (float*)alloc((size_t)MTOT * HVn * 4);
  float* betab   = (float*)alloc((size_t)MTOT * HVn * 4);
  float* gbuf    = (float*)alloc((size_t)MTOT * HVn * 4);
  __bf16* hbf    = (__bf16*)alloc((size_t)MTOT * Dd * 2);
  __bf16* wqkvbf = (__bf16*)alloc((size_t)Dd * CONVD * 2);
  __bf16* wzbf   = (__bf16*)alloc((size_t)Dd * VALD * 2);
  __bf16* woutbf = (__bf16*)alloc((size_t)VALD * Dd * 2);
  __bf16* nbf    = (__bf16*)alloc((size_t)MTOT * VALD * 2);

  auto cdiv = [](long a, long b) { return (int)((a + b - 1) / b); };

  // bf16 casts for GEMM operands
  k_cast_bf16<<<cdiv((long)MTOT * Dd, 256), 256, 0, stream>>>(hidden, hbf, (long)MTOT * Dd);
  k_cast_bf16<<<cdiv((long)Dd * CONVD, 256), 256, 0, stream>>>(Wqkv, wqkvbf, (long)Dd * CONVD);
  k_cast_bf16<<<cdiv((long)Dd * VALD, 256), 256, 0, stream>>>(Wz, wzbf, (long)Dd * VALD);
  k_cast_bf16<<<cdiv((long)VALD * Dd, 256), 256, 0, stream>>>(Wout, woutbf, (long)VALD * Dd);

  // mixed = hidden @ W_qkv ; z = hidden @ W_z (bf16 WMMA, async staging)
  k_gemm_bf16<<<dim3(CONVD / 128, MTOT / 128), 256, 0, stream>>>(hbf, wqkvbf, mixed, MTOT, CONVD, Dd);
  k_gemm_bf16<<<dim3(VALD / 128, MTOT / 128), 256, 0, stream>>>(hbf, wzbf, zbuf, MTOT, VALD, Dd);

  // b/a projections (N=32 each)
  k_ba_proj<<<cdiv((long)MTOT * 64, 256), 256, 0, stream>>>(hidden, Wb, Wa, bp, ap);

  // conv + silu + split; q/k l2 norm; beta/g
  k_conv_silu<<<cdiv((long)MTOT * CONVD, 256), 256, 0, stream>>>(mixed, cw, qbuf, kbuf, vbuf);
  k_qk_norm<<<(2 * MTOT * HK) / 8, 256, 0, stream>>>(qbuf, kbuf);
  k_beta_g<<<cdiv((long)MTOT * HVn, 256), 256, 0, stream>>>(bp, ap, dtb, Alog, betab, gbuf);

  // chunked gated delta rule (fp32 WMMA + TDM prefetch), 64 head-streams
  k_delta_core<<<Bb * HVn, 256, 0, stream>>>(qbuf, kbuf, vbuf, betab, gbuf, corebuf);

  // gated RMSNorm -> bf16, then output GEMM
  k_gate_norm<<<(MTOT * HVn) / 8, 256, 0, stream>>>(corebuf, zbuf, nw, nbf);
  k_gemm_bf16<<<dim3(Dd / 128, MTOT / 128), 256, 0, stream>>>(nbf, woutbf, out, MTOT, Dd, VALD);
}
